// SS2D_20444044329276
// MI455X (gfx1250) — matrix operand order repaired and checked by reading, not compiled, not run on hardware
//
#include <hip/hip_runtime.h>
#include <hip/hip_bf16.h>
#include <math.h>

// ---------------------------------------------------------------------------
// SS2D forward for MI455X (gfx1250, wave32, WMMA + async LDS DMA).
// B=1, C=64, H=W=64, depth=4, d_inner=128, d_state=16, dt_rank=4, conv K=4.
// 764 sequences (4x64 axis runs + 4x127 diagonal runs), each L=64.
// prep_weights (f32->f16) -> mamba_stack (async gather + 4 layers + async
// writeback) -> fusion (fused scatter + attention + softmax + residual).
// ---------------------------------------------------------------------------

typedef __attribute__((ext_vector_type(16))) _Float16 v16h;
typedef __attribute__((ext_vector_type(8)))  _Float16 v8h;
typedef __attribute__((ext_vector_type(8)))  float    v8f;

#define NSEQ   764
#define LSEQ   64
#define CH     64
#define DIN    128
#define DSTATE 16

// LDS strides
#define LDX 66      // floats
#define LDU 130     // floats
#define LDD 52      // floats
#define LDA16 136   // halves (16B-aligned rows: 136*2 = 272 = 17*16)

// f16 weight pack offsets (in halves) within one param set
#define W16_INW   0
#define W16_XPROJ 65536          // 4*256*64
#define W16_DT    83968          // + 4*36*128
#define W16_OUT   86016          // + 4*128*4
#define W16_SET   118784         // + 4*64*128  (multiple of 128 -> 256B align)

struct MParams {
  const float *in_w, *conv_w, *conv_b, *xproj_w, *dt_w,
              *dt_b, *A_log, *Dw, *out_w, *rms_w;
};
struct AllParams { MParams ps[4]; };

// ---------------- LDS raw-offset helper (for async-to-LDS asm) -------------
typedef __attribute__((address_space(3))) const void lds_cvoid;
__device__ __forceinline__ unsigned lds_addr32(const void* p) {
  return (unsigned)(unsigned long long)(lds_cvoid*)p;
}

// ---------------- WMMA fragment helpers (A/B: contiguous k = 16h + s; C/D: ISA layout) ---------------
// Stage A[64][Kdim] (f32, stride lda) -> f16 LDS (stride LDA16), zero-pad
// columns >= kmax. One conversion per element, reused by all N-tiles.
__device__ __forceinline__ void stage_a16(const float* A, int lda, int kshift,
                                          int kmax, _Float16* A16, int tid) {
  int Kdim = 1 << kshift;
  for (int i = tid; i < (64 << kshift); i += 256) {
    int m = i >> kshift, k = i & (Kdim - 1);
    A16[m * LDA16 + k] = (k < kmax) ? (_Float16)A[m * lda + k] : (_Float16)0.f;
  }
}

// A 16x32 f16: lanes 0-15 -> M=lane, K = h ; lanes 16-31 add K+16.
// Two aligned 16B LDS loads per fragment, no conversions.
__device__ __forceinline__ v16h load_a_h(const _Float16* A16, int m0, int k0,
                                         int lane) {
  const _Float16* row = A16 + (m0 + (lane & 15)) * LDA16
                             + k0 + (lane & 16);
  v8h lo = *reinterpret_cast<const v8h*>(row);
  v8h hi = *reinterpret_cast<const v8h*>(row + 8);
  return __builtin_shufflevector(lo, hi, 0, 1, 2, 3, 4, 5, 6, 7,
                                         8, 9, 10, 11, 12, 13, 14, 15);
}

// B 32x16 f16 from pre-packed f16 weights W[N][K] (y = x * W^T):
// lanes 0-15: N=lane, K=k0+h ; lanes 16-31: K += 16.
__device__ __forceinline__ v16h load_b_h(const _Float16* W, int ldw, int n0,
                                         int k0, int kmax, int nmax, int lane) {
  int n  = n0 + (lane & 15);
  int kb = k0 + (lane & 16);
  if (n < nmax && (kb + 16) <= kmax)       // fast path: 32B aligned vector load
    return *reinterpret_cast<const v16h*>(W + n * ldw + kb);
  v16h b;
#pragma unroll
  for (int h = 0; h < 16; ++h) {
    int k = kb + h;
    b[h] = (n < nmax && k < kmax) ? W[n * ldw + k] : (_Float16)0.f;
  }
  return b;
}

// C/D 16x16 f32: lanes 0-15: N=lane, VGPR r -> M=r ; lanes 16-31 -> M=r+8.
__device__ __forceinline__ void store_c_f(float* S, int ldc, int m0, int n0,
                                          int lane, v8f c) {
  int n  = n0 + (lane & 15);
  int mb = m0 + ((lane & 16) >> 1);
#pragma unroll
  for (int r = 0; r < 8; ++r) S[(mb + r) * ldc + n] = c[r];
}

// C[M=64,N] = A16[64,K] * W[N,K]^T. Tile counts are always even, so each
// wave owns pairs of adjacent tiles (2p, 2p+1): two independent accumulator
// chains hide the WMMA->WMMA accumulator hazard (ISA 7.12.1) with no
// divergent branches; for even nt the pair shares its A fragments.
// `wave` must be an SGPR value (readfirstlane) so loop control stays scalar.
__device__ __forceinline__ void gemm_wmma(const _Float16* A16, int Kpad,
                                          const _Float16* W, int ldw, int kmax,
                                          int nmax, float* C, int ldc, int N,
                                          int wave, int lane) {
  int nt = N >> 4;
  int npairs = (4 * nt) >> 1;
  for (int p = wave; p < npairs; p += 8) {
    int t0 = 2 * p, t1 = 2 * p + 1;
    int m0 = (t0 / nt) << 4, n0 = (t0 % nt) << 4;
    int m1 = (t1 / nt) << 4, n1 = (t1 % nt) << 4;
    v8f c0 = {}, c1 = {};
    for (int k0 = 0; k0 < Kpad; k0 += 32) {
      v16h a0 = load_a_h(A16, m0, k0, lane);
      v16h b0 = load_b_h(W, ldw, n0, k0, kmax, nmax, lane);
      c0 = __builtin_amdgcn_wmma_f32_16x16x32_f16(
               false, a0, false, b0, (short)0, c0, false, false);
      v16h a1 = load_a_h(A16, m1, k0, lane);
      v16h b1 = load_b_h(W, ldw, n1, k0, kmax, nmax, lane);
      c1 = __builtin_amdgcn_wmma_f32_16x16x32_f16(
               false, a1, false, b1, (short)0, c1, false, false);
    }
    store_c_f(C, ldc, m0, n0, lane, c0);
    store_c_f(C, ldc, m1, n1, lane, c1);
  }
}

// ------------------------- sequence index helpers --------------------------
__device__ __forceinline__ int seq_run(int s, int& sl) {
  if (s < 256) { sl = s & 63;  return s >> 6; }
  if (s < 383) { sl = s - 256; return 4; }
  if (s < 510) { sl = s - 383; return 5; }
  if (s < 637) { sl = s - 510; return 6; }
  sl = s - 637; return 7;
}

// (run, diag/row id sl, position t) -> (j,k) in the HxW image; false = padded
__device__ __forceinline__ bool gather_jk(int run, int sl, int t, int& j, int& k) {
  switch (run) {
    case 0: j = sl;     k = t;      return true;
    case 1: j = sl;     k = 63 - t; return true;
    case 2: j = t;      k = sl;     return true;
    case 3: j = 63 - t; k = sl;     return true;
    case 4: { int j0 = max(0, sl - 63), L = min(63, sl) - j0 + 1;
              if (t >= L) return false; j = j0 + t;         k = sl - j; return true; }
    case 5: { int j0 = max(0, sl - 63), L = min(63, sl) - j0 + 1;
              if (t >= L) return false; j = j0 + L - 1 - t; k = sl - j; return true; }
    case 6: { int j0 = max(0, 63 - sl), L = min(63, 126 - sl) - j0 + 1;
              if (t >= L) return false; j = j0 + t;         k = sl - (63 - j); return true; }
    default:{ int j0 = max(0, 63 - sl), L = min(63, 126 - sl) - j0 + 1;
              if (t >= L) return false; j = j0 + L - 1 - t; k = sl - (63 - j); return true; }
  }
}

__device__ __forceinline__ float siluf(float x) { return x / (1.f + __expf(-x)); }

// ---------------------------------------------------------------------------
// Kernel 0: pre-convert GEMM weights to f16 (B-operand fast path).
// ---------------------------------------------------------------------------
__global__ __launch_bounds__(256) void prep_weights(AllParams ap,
                                                    _Float16* __restrict__ wbuf) {
  int idx = blockIdx.x * 256 + threadIdx.x;
  if (idx >= 4 * W16_SET) return;
  int set = idx / W16_SET, r = idx - set * W16_SET;
  const MParams P = ap.ps[set];
  float v;
  if      (r < W16_XPROJ) v = P.in_w  [r];
  else if (r < W16_DT)    v = P.xproj_w[r - W16_XPROJ];
  else if (r < W16_OUT)   v = P.dt_w  [r - W16_DT];
  else                    v = P.out_w [r - W16_OUT];
  wbuf[idx] = (_Float16)v;
}

// ---------------------------------------------------------------------------
// Kernel 1: fused gather (async DMA to LDS) + depth-4 Mamba stack + async
// writeback. One workgroup (8 wave32) per sequence; ~147 KB LDS.
// ---------------------------------------------------------------------------
__global__ __launch_bounds__(256) void mamba_stack_kernel(
    const float* __restrict__ x, float* __restrict__ seqbuf,
    const _Float16* __restrict__ wbuf, AllParams ap) {
  __shared__ float xin  [LSEQ * LDX];    // layer input / out_proj dest
  __shared__ float bufU [LSEQ * LDU];    // u -> delta -> y_final
  __shared__ float bufUc[LSEQ * LDU];    // uc
  __shared__ float bufDbl[LSEQ * LDD];   // [dt(4) | B(16) | C(16) | pad]
  __shared__ float bufZ [LSEQ * LDU];    // z
  __shared__ _Float16 aStage[LSEQ * LDA16];  // f16 A-operand staging

  int tid = threadIdx.x, lane = tid & 31;
  int wave = __builtin_amdgcn_readfirstlane(tid >> 5);   // SGPR wave index
  int s = blockIdx.x, sl;
  int run = seq_run(s, sl);
  int set = run >> 1;
  const MParams P = ap.ps[set];
  const _Float16* wb = wbuf + (size_t)set * W16_SET;

  // ---- stage input sequence: per-lane async global->LDS DMA (diag-aware) --
  for (int i = tid; i < LSEQ * CH; i += 256) {
    int t = i >> 6, c = i & 63;
    int j, k;
    bool valid = gather_jk(run, sl, t, j, k);
    float* dstp = &xin[t * LDX + c];
    if (valid) {
      unsigned lo = lds_addr32(dstp);
      unsigned long long ga =
          (unsigned long long)(const void*)(x + c * 4096 + j * 64 + k);
      asm volatile("global_load_async_to_lds_b32 %0, %1, off"
                   :: "v"(lo), "v"(ga) : "memory");
    } else {
      *dstp = 0.f;
    }
  }
  asm volatile("s_wait_asynccnt 0" ::: "memory");
  __syncthreads();

  for (int layer = 0; layer < 4; ++layer) {
    const _Float16* in_w16  = wb + W16_INW   + layer * 256 * 64;
    const _Float16* xproj16 = wb + W16_XPROJ + layer * 36 * DIN;
    const _Float16* dt16    = wb + W16_DT    + layer * DIN * 4;
    const _Float16* out16   = wb + W16_OUT   + layer * 64 * DIN;
    const float* conv_w = P.conv_w + layer * DIN * 4;
    const float* conv_b = P.conv_b + layer * DIN;
    const float* dt_b   = P.dt_b   + layer * DIN;
    const float* A_log  = P.A_log  + layer * DIN * DSTATE;
    const float* Dw     = P.Dw     + layer * DIN;
    const float* rms_w  = P.rms_w  + layer * 64;

    // 1) stage xin once; u and z GEMMs share the A operand
    stage_a16(xin, LDX, 6, 64, aStage, tid);
    __syncthreads();
    gemm_wmma(aStage, 64, in_w16,            64, 64, 128, bufU, LDU, 128, wave, lane);
    gemm_wmma(aStage, 64, in_w16 + 128 * 64, 64, 64, 128, bufZ, LDU, 128, wave, lane);
    __syncthreads();

    // 2) uc = silu(causal conv(u) + b): branchless register sliding window
    if (tid < DIN) {
      int d = tid;
      float cw0 = conv_w[d * 4 + 0], cw1 = conv_w[d * 4 + 1];
      float cw2 = conv_w[d * 4 + 2], cw3 = conv_w[d * 4 + 3];
      float cb = conv_b[d];
      float um1 = 0.f, um2 = 0.f, um3 = 0.f;
      for (int t = 0; t < LSEQ; ++t) {
        float u = bufU[t * LDU + d];
        float acc = cb + cw3 * u + cw2 * um1 + cw1 * um2 + cw0 * um3;
        bufUc[t * LDU + d] = siluf(acc);
        um3 = um2; um2 = um1; um1 = u;
      }
    }
    __syncthreads();

    // 3) dbl = uc @ xproj_w.T                  (M64,N36->48,K128)
    stage_a16(bufUc, LDU, 7, 128, aStage, tid);
    __syncthreads();
    gemm_wmma(aStage, 128, xproj16, 128, 128, 36, bufDbl, LDD, 48, wave, lane);
    __syncthreads();

    // 4) delta_raw = dbl[:, :4] @ dt_w.T       (M64,N128,K4 zero-padded)
    stage_a16(bufDbl, LDD, 5, 4, aStage, tid);
    __syncthreads();
    gemm_wmma(aStage, 32, dt16, 4, 4, 128, bufU, LDU, 128, wave, lane);
    __syncthreads();

    // 5) delta = softplus(delta_raw + dt_b)
    for (int i = tid; i < LSEQ * DIN; i += 256) {
      int t = i >> 7, d = i & 127;
      float v = bufU[t * LDU + d] + dt_b[d];
      bufU[t * LDU + d] = (v > 20.f) ? v : log1pf(__expf(v));
    }
    __syncthreads();

    // 6) selective scan; thread d owns 16 f32 states in registers.
    //    Fuses + u*D and * silu(z); writes y_final into bufU in place
    //    (each element is read then written by its owning thread).
    if (tid < DIN) {
      int d = tid;
      float Arow[DSTATE], h[DSTATE];
#pragma unroll
      for (int ss = 0; ss < DSTATE; ++ss) {
        Arow[ss] = -__expf(A_log[d * DSTATE + ss]);
        h[ss] = 0.f;
      }
      float Dv = Dw[d];
      for (int t = 0; t < LSEQ; ++t) {
        float dt = bufU [t * LDU + d];
        float u  = bufUc[t * LDU + d];
        float zz = bufZ [t * LDU + d];
        float du = dt * u;
        const float4* Bp = reinterpret_cast<const float4*>(&bufDbl[t * LDD + 4]);
        const float4* Cp = reinterpret_cast<const float4*>(&bufDbl[t * LDD + 20]);
        float y = 0.f;
#pragma unroll
        for (int q = 0; q < 4; ++q) {
          float4 B4 = Bp[q], C4 = Cp[q];
          float e0 = __expf(dt * Arow[q * 4 + 0]);
          float e1 = __expf(dt * Arow[q * 4 + 1]);
          float e2 = __expf(dt * Arow[q * 4 + 2]);
          float e3 = __expf(dt * Arow[q * 4 + 3]);
          h[q * 4 + 0] = h[q * 4 + 0] * e0 + du * B4.x;
          h[q * 4 + 1] = h[q * 4 + 1] * e1 + du * B4.y;
          h[q * 4 + 2] = h[q * 4 + 2] * e2 + du * B4.z;
          h[q * 4 + 3] = h[q * 4 + 3] * e3 + du * B4.w;
          y += h[q * 4 + 0] * C4.x + h[q * 4 + 1] * C4.y
             + h[q * 4 + 2] * C4.z + h[q * 4 + 3] * C4.w;
        }
        bufU[t * LDU + d] = (y + u * Dv) * siluf(zz);
      }
    }
    __syncthreads();

    // 7) out = y_final @ out_w.T               (M64,N64,K128) -> xin
    stage_a16(bufU, LDU, 7, 128, aStage, tid);
    __syncthreads();
    gemm_wmma(aStage, 128, out16, 128, 128, 64, xin, LDX, 64, wave, lane);
    __syncthreads();

    // 8) RMSNorm over channels, in place
    if (tid < LSEQ) {
      int t = tid;
      float ss = 0.f;
      for (int c = 0; c < CH; ++c) { float v = xin[t * LDX + c]; ss += v * v; }
      float r = rsqrtf(ss * (1.f / CH) + 1e-5f);
      for (int c = 0; c < CH; ++c) xin[t * LDX + c] *= r * rms_w[c];
    }
    __syncthreads();
  }

  // ---- async writeback LDS -> seqbuf --------------------------------------
  float* gseq = seqbuf + (size_t)s * (LSEQ * CH);
  for (int i = tid; i < LSEQ * CH; i += 256) {
    unsigned lo = lds_addr32(&xin[(i >> 6) * LDX + (i & 63)]);
    unsigned long long ga = (unsigned long long)(const void*)(gseq + i);
    asm volatile("global_store_async_from_lds_b32 %0, %1, off"
                 :: "v"(ga), "v"(lo) : "memory");
  }
  asm volatile("s_wait_asynccnt 0" ::: "memory");
}

// ---------------------------------------------------------------------------
// Kernel 2: fused scatter + attention fusion + softmax + residual.
// One 64-thread block (2 wave32) per pixel; reads seqbuf via inverse maps.
// ---------------------------------------------------------------------------
__global__ __launch_bounds__(64) void fusion_kernel(
    const float* __restrict__ x, const float* __restrict__ seqbuf,
    const float* __restrict__ w1, const float* __restrict__ b1,
    const float* __restrict__ w2, const float* __restrict__ b2,
    float* __restrict__ out) {
  __shared__ float hv[64], a1[16], a2[8], wsm[8];
  int p = blockIdx.x, c = threadIdx.x;
  int j = p >> 6, k = p & 63;

  // inverse maps: per direction the (sequence, position) holding pixel (j,k)
  int i1 = j + k,       j01 = max(0, i1 - 63), L1 = min(63, i1) - j01 + 1;
  int i2 = k + 63 - j,  j02 = max(0, 63 - i2), L2 = min(63, 126 - i2) - j02 + 1;
  int s_[8] = { j, 64 + j, 128 + k, 192 + k,
                256 + i1, 383 + i1, 510 + i2, 637 + i2 };
  int t_[8] = { k, 63 - k, j, 63 - j,
                j - j01, L1 - 1 - (j - j01), j - j02, L2 - 1 - (j - j02) };

  hv[c] = seqbuf[(size_t)s_[0] * 4096 + t_[0] * 64 + c];   // hr
  __syncthreads();
  if (c < 16) {
    float s = b1[c];
    for (int i = 0; i < 64; ++i) s += w1[c * 64 + i] * hv[i];
    a1[c] = fmaxf(s, 0.f);
  }
  __syncthreads();
  if (c < 8) {
    float s = b2[c];
    for (int i = 0; i < 16; ++i) s += w2[c * 16 + i] * a1[i];
    a2[c] = s;
  }
  __syncthreads();
  if (c == 0) {
    float m = a2[0];
    for (int i = 1; i < 8; ++i) m = fmaxf(m, a2[i]);
    float e[8], den = 0.f;
    for (int i = 0; i < 8; ++i) { e[i] = __expf(a2[i] - m); den += e[i]; }
    float inv = 1.f / den;
    for (int i = 0; i < 8; ++i) wsm[i] = e[i] * inv;
  }
  __syncthreads();
  float acc = x[c * 4096 + p];                              // residual
#pragma unroll
  for (int i = 0; i < 8; ++i)
    acc += wsm[i] * seqbuf[(size_t)s_[i] * 4096 + t_[i] * 64 + c];
  out[c * 4096 + p] = acc;
}

// ---------------------------------------------------------------------------
extern "C" void kernel_launch(void* const* d_in, const int* in_sizes, int n_in,
                              void* d_out, int out_size, void* d_ws, size_t ws_size,
                              hipStream_t stream) {
  (void)in_sizes; (void)n_in; (void)out_size; (void)ws_size;
  // d_in order (setup_inputs insertion order):
  // 0: x ; 1-10: params_w ; 11-20: params_h ; 21-30: params_d1 ;
  // 31-40: params_d2 ; 41-44: attn {w1,b1,w2,b2}
  const float* x = (const float*)d_in[0];
  AllParams ap;
  for (int g = 0; g < 4; ++g) {
    void* const* b = d_in + 1 + g * 10;
    ap.ps[g].in_w    = (const float*)b[0];
    ap.ps[g].conv_w  = (const float*)b[1];
    ap.ps[g].conv_b  = (const float*)b[2];
    ap.ps[g].xproj_w = (const float*)b[3];
    ap.ps[g].dt_w    = (const float*)b[4];
    ap.ps[g].dt_b    = (const float*)b[5];
    ap.ps[g].A_log   = (const float*)b[6];
    ap.ps[g].Dw      = (const float*)b[7];
    ap.ps[g].out_w   = (const float*)b[8];
    ap.ps[g].rms_w   = (const float*)b[9];
  }
  const float* w1 = (const float*)d_in[41];
  const float* b1 = (const float*)d_in[42];
  const float* w2 = (const float*)d_in[43];
  const float* b2 = (const float*)d_in[44];

  // workspace: seqbuf 764*4096 f32 (12.5MB), then f16 weight pack (~0.95MB)
  float*     seqbuf = (float*)d_ws;
  _Float16*  wbuf   = (_Float16*)((char*)d_ws + (size_t)NSEQ * 4096 * 4);

  prep_weights      <<<(4 * W16_SET + 255) / 256, 256, 0, stream>>>(ap, wbuf);
  mamba_stack_kernel<<<NSEQ, 256, 0, stream>>>(x, seqbuf, wbuf, ap);
  fusion_kernel     <<<4096,  64, 0, stream>>>(x, seqbuf, w1, b1, w2, b2,
                                               (float*)d_out);
}
